// UMT_MVSNet_V1_46797963657723
// MI455X (gfx1250) — matrix operand order repaired and checked by reading, not compiled, not run on hardware
//
#include <hip/hip_runtime.h>
#include <math.h>

// ---------------------------------------------------------------------------
// UMT-MVSNet forward for gfx1250 (MI455X).  Conv layers use implicit-GEMM via
// v_wmma_f32_16x16x32_f16 (wave32, 16(cout) x 32(pixel) tile per wave, two B
// fragments sharing one A-fragment load).  Weights are pre-packed into the
// A-fragment VGPR layout so A loads are coalesced 32B/lane v16h loads.
// Conv kernel is templated on KSZ so im2col index math uses constant
// divisors (mul-shift) instead of runtime division.
// ---------------------------------------------------------------------------

typedef _Float16 half_t;
typedef __attribute__((ext_vector_type(16))) _Float16 v16h;
typedef __attribute__((ext_vector_type(8)))  float    v8f;

#define EW_BLK 256

__device__ __forceinline__ float umt_sig(float x) { return 1.f / (1.f + expf(-x)); }

// ------------------------------ elementwise --------------------------------

__global__ void umt_zero_kernel(float* __restrict__ p, int n) {
  int i = blockIdx.x * blockDim.x + threadIdx.x;
  if (i < n) p[i] = 0.f;
}

__global__ void umt_copy_kernel(float* __restrict__ dst, const float* __restrict__ src, int n) {
  int i = blockIdx.x * blockDim.x + threadIdx.x;
  if (i < n) dst[i] = src[i];
}

__global__ void umt_add_relu_kernel(float* __restrict__ y, const float* __restrict__ a,
                                    const float* __restrict__ b, int n) {
  int i = blockIdx.x * blockDim.x + threadIdx.x;
  if (i < n) { float v = a[i] + b[i]; y[i] = v > 0.f ? v : 0.f; }
}

__global__ void umt_sigmoid_kernel(float* __restrict__ x, int n) {
  int i = blockIdx.x * blockDim.x + threadIdx.x;
  if (i < n) x[i] = umt_sig(x[i]);
}

__global__ void umt_scale_kernel(float* __restrict__ y, const float* __restrict__ x, float s, int n) {
  int i = blockIdx.x * blockDim.x + threadIdx.x;
  if (i < n) y[i] = x[i] * s;
}

__global__ void umt_acc_kernel(float* __restrict__ acc, const float* __restrict__ wv,
                               const float* __restrict__ rw, int n, int HW, int first) {
  int i = blockIdx.x * blockDim.x + threadIdx.x;
  if (i >= n) return;
  float v = (rw[i % HW] + 1.f) * wv[i];
  acc[i] = first ? v : (acc[i] + v);
}

__global__ void umt_maxpool2_kernel(const float* __restrict__ x, float* __restrict__ y,
                                    int C, int H, int W) {
  int Ho = H >> 1, Wo = W >> 1;
  int n = C * Ho * Wo;
  int i = blockIdx.x * blockDim.x + threadIdx.x;
  if (i >= n) return;
  int c = i / (Ho * Wo);
  int r = i % (Ho * Wo);
  int oy = r / Wo, ox = r % Wo;
  const float* xb = x + ((size_t)c * H + 2 * oy) * W + 2 * ox;
  y[i] = fmaxf(fmaxf(xb[0], xb[1]), fmaxf(xb[W], xb[W + 1]));
}

__global__ void umt_up2_kernel(const float* __restrict__ x, float* __restrict__ y,
                               int C, int H, int W) {
  int Ho = H * 2, Wo = W * 2;
  int n = C * Ho * Wo;
  int i = blockIdx.x * blockDim.x + threadIdx.x;
  if (i >= n) return;
  int c = i / (Ho * Wo);
  int r = i % (Ho * Wo);
  int oy = r / Wo, ox = r % Wo;
  y[i] = x[((size_t)c * H + (oy >> 1)) * W + (ox >> 1)];
}

__global__ void umt_lstm_kernel(const float* __restrict__ z, float* __restrict__ h,
                                float* __restrict__ c, int Cg, int HW) {
  int n = Cg * HW;
  int i = blockIdx.x * blockDim.x + threadIdx.x;
  if (i >= n) return;
  size_t off = (size_t)Cg * HW;
  float zi = z[i];
  float zf = z[i + off];
  float zo = z[i + 2 * off];
  float zg = z[i + 3 * off];
  float c2 = umt_sig(zf) * c[i] + umt_sig(zi) * tanhf(zg);
  h[i] = umt_sig(zo) * tanhf(c2);
  c[i] = c2;
}

// ---------------------------- group norm + relu ----------------------------
// one block (256 thr) per group; two-pass with shared reduction; in-place ok.
__global__ __launch_bounds__(256)
void umt_gn_relu_kernel(const float* __restrict__ x, float* __restrict__ y,
                        int C, int HW, int groups, float eps) {
  __shared__ float ssum[256];
  __shared__ float ssq[256];
  int g = blockIdx.x;
  int cpg = C / groups;
  size_t n = (size_t)cpg * HW;
  const float* xg = x + (size_t)g * cpg * HW;
  float*       yg = y + (size_t)g * cpg * HW;
  float s = 0.f, sq = 0.f;
  for (size_t i = threadIdx.x; i < n; i += 256) { float v = xg[i]; s += v; sq += v * v; }
  ssum[threadIdx.x] = s; ssq[threadIdx.x] = sq;
  __syncthreads();
  for (int st = 128; st > 0; st >>= 1) {
    if ((int)threadIdx.x < st) { ssum[threadIdx.x] += ssum[threadIdx.x + st];
                                 ssq[threadIdx.x]  += ssq[threadIdx.x + st]; }
    __syncthreads();
  }
  float mean = ssum[0] / (float)n;
  float var  = ssq[0] / (float)n - mean * mean;
  float sc   = rsqrtf(var + eps);
  for (size_t i = threadIdx.x; i < n; i += 256) {
    float v = (xg[i] - mean) * sc;
    yg[i] = v > 0.f ? v : 0.f;
  }
}

// ----------------------------- weight packing ------------------------------
// Pack f32 OIHW weights into A-fragment layout (16x32 f16 per tile-chunk):
// lane L: row M = L&15, half = L>>4;  vgpr i holds K pair at
//   kbase = kc*32 + (i>>2)*16 + half*8 + (i&3)*2.
__global__ void umt_pack_w_kernel(const float* __restrict__ w, half_t* __restrict__ pw,
                                  int Cin, int Cout, int ksz) {
  int K  = Cin * ksz * ksz;
  int KC = (K + 31) >> 5;
  int CT = (Cout + 15) >> 4;
  int total = CT * KC * 256;
  int t = blockIdx.x * blockDim.x + threadIdx.x;
  if (t >= total) return;
  int i    = t & 7;
  int lane = (t >> 3) & 31;
  int chnk = t >> 8;
  int kc = chnk % KC;
  int ct = chnk / KC;
  int M = lane & 15, hf = lane >> 4;
  int kb = kc * 32 + (i >> 2) * 16 + hf * 8 + (i & 3) * 2;
  int co = ct * 16 + M;
  float v0 = 0.f, v1 = 0.f;
  if (co < Cout) {
    if (kb     < K) v0 = w[(size_t)co * K + kb];
    if (kb + 1 < K) v1 = w[(size_t)co * K + kb + 1];
  }
  size_t base = (((size_t)ct * KC + kc) * 32 + lane) * 16 + 2 * i;
  pw[base]     = (half_t)v0;
  pw[base + 1] = (half_t)v1;
}

// ------------------------------- WMMA conv ---------------------------------
// One wave computes a 16(cout) x 32(pixel) tile: two 16x16 WMMAs sharing one
// A-fragment load.  4 waves / block.  B fragment element e of lane L maps to
// K = kc*32 + (L>>4)*8 + (e&7) + (e>>3)*16, column N = L&15.
template <int KSZ>
__global__ __launch_bounds__(128)
void umt_conv_wmma_kernel(const float* __restrict__ x, const half_t* __restrict__ pw,
                          const float* __restrict__ bias, float* __restrict__ y,
                          int Cin, int Cout, int Hin, int Win, int Hout, int Wout,
                          int stride) {
  constexpr int KK  = KSZ * KSZ;
  constexpr int PAD = KSZ / 2;
  const int K    = Cin * KK;
  const int KC   = (K + 31) >> 5;
  const int npix = Hout * Wout;
  const int lane = threadIdx.x & 31;
  const int wvid = threadIdx.x >> 5;
  const int pt   = blockIdx.x * 4 + wvid;   // 32-pixel tile (wave-uniform)
  const int ct   = blockIdx.y;              // cout tile
  if (pt * 32 >= npix) return;              // wave-uniform exit: EXEC stays full

  const int col = lane & 15;
  const int hf  = lane >> 4;
  const int p0  = pt * 32 + col;
  const int p1  = p0 + 16;
  const bool pv0 = p0 < npix;
  const bool pv1 = p1 < npix;
  const int oh0 = p0 / Wout, ow0 = p0 - oh0 * Wout;
  const int oh1 = p1 / Wout, ow1 = p1 - oh1 * Wout;
  const int ihb0 = oh0 * stride - PAD, iwb0 = ow0 * stride - PAD;
  const int ihb1 = oh1 * stride - PAD, iwb1 = ow1 * stride - PAD;
  const int kofs = hf * 8;

  v8f acc0 = {}, acc1 = {};
  for (int kc = 0; kc < KC; ++kc) {
    // A: coalesced packed-weight fragment (32 B per lane)
    const half_t* ap = pw + (((size_t)ct * KC + kc) * 32 + lane) * 16;
    v16h A = *(const v16h*)ap;
    // speculative prefetch of next chunk's A fragment (global_prefetch_b8)
    __builtin_prefetch((const void*)(ap + 512), 0, 1);

    // B0/B1: per-lane im2col gather + f32->f16 (constant-divisor decode)
    v16h B0, B1;
    const int kbase = kc * 32 + kofs;
#pragma unroll
    for (int e = 0; e < 16; ++e) {
      int k = kbase + (e & 7) + ((e >> 3) << 4);
      float b0 = 0.f, b1 = 0.f;
      if (k < K) {
        int ci = k / KK;            // KK is a compile-time constant
        int r  = k - ci * KK;
        int dh = r / KSZ;
        int dw = r - dh * KSZ;
        const float* xc = x + (size_t)ci * Hin * Win;
        int ih0 = ihb0 + dh, iw0 = iwb0 + dw;
        int ih1 = ihb1 + dh, iw1 = iwb1 + dw;
        if (pv0 && ih0 >= 0 && ih0 < Hin && iw0 >= 0 && iw0 < Win)
          b0 = xc[ih0 * Win + iw0];
        if (pv1 && ih1 >= 0 && ih1 < Hin && iw1 >= 0 && iw1 < Win)
          b1 = xc[ih1 * Win + iw1];
      }
      B0[e] = (_Float16)b0;
      B1[e] = (_Float16)b1;
    }
    acc0 = __builtin_amdgcn_wmma_f32_16x16x32_f16(false, A, false, B0,
                                                  (short)0, acc0, false, false);
    acc1 = __builtin_amdgcn_wmma_f32_16x16x32_f16(false, A, false, B1,
                                                  (short)0, acc1, false, false);
  }
  // C layout: vgpr r -> M = r + 8*hf, N = lane&15
#pragma unroll
  for (int r = 0; r < 8; ++r) {
    int co = ct * 16 + r + hf * 8;
    if (co < Cout) {
      float b = bias[co];
      if (pv0) y[(size_t)co * npix + p0] = acc0[r] + b;
      if (pv1) y[(size_t)co * npix + p1] = acc1[r] + b;
    }
  }
}

// ------------------------- homography warp + (.-ref)^2 ---------------------
__global__ void umt_warp_sq_kernel(const float* __restrict__ src, const float* __restrict__ ref,
                                   const float* __restrict__ proj, const float* __restrict__ depthv,
                                   int didx, float* __restrict__ wv, int C, int Hf, int Wf) {
  int HW = Hf * Wf;
  int p = blockIdx.x * blockDim.x + threadIdx.x;
  if (p >= HW) return;
  float depth = depthv[didx];
  float xs = (float)(p % Wf), ys = (float)(p / Wf);
  float px = (proj[0] * xs + proj[1] * ys + proj[2])  * depth + proj[3];
  float py = (proj[4] * xs + proj[5] * ys + proj[6])  * depth + proj[7];
  float pz = (proj[8] * xs + proj[9] * ys + proj[10]) * depth + proj[11];
  float z  = (fabsf(pz) < 1e-6f) ? 1e-6f : pz;
  float fx = px / z, fy = py / z;
  float x0f = floorf(fx), y0f = floorf(fy);
  float wx = fx - x0f, wy = fy - y0f;
  int x0 = (int)x0f, y0 = (int)y0f;
  int xs4[4] = {x0, x0 + 1, x0, x0 + 1};
  int ys4[4] = {y0, y0, y0 + 1, y0 + 1};
  float ww[4] = {(1.f - wx) * (1.f - wy), wx * (1.f - wy), (1.f - wx) * wy, wx * wy};
  int idx[4]; float wgt[4];
#pragma unroll
  for (int t = 0; t < 4; ++t) {
    bool valid = xs4[t] >= 0 && xs4[t] <= Wf - 1 && ys4[t] >= 0 && ys4[t] <= Hf - 1;
    int xc = xs4[t] < 0 ? 0 : (xs4[t] > Wf - 1 ? Wf - 1 : xs4[t]);
    int yc = ys4[t] < 0 ? 0 : (ys4[t] > Hf - 1 ? Hf - 1 : ys4[t]);
    idx[t] = yc * Wf + xc;
    wgt[t] = valid ? ww[t] : 0.f;
  }
  for (int c = 0; c < C; ++c) {
    const float* sc = src + (size_t)c * HW;
    float v = wgt[0] * sc[idx[0]] + wgt[1] * sc[idx[1]] +
              wgt[2] * sc[idx[2]] + wgt[3] * sc[idx[3]];
    float d = v - ref[(size_t)c * HW + p];
    wv[(size_t)c * HW + p] = d * d;
  }
}

// --------------------- proj_v @ inv(proj_0) (1 thread) ---------------------
__global__ void umt_proj_prep_kernel(const float* __restrict__ pm, float* __restrict__ out, int V) {
  if (blockIdx.x != 0 || threadIdx.x != 0) return;
  float a[4][8];
  for (int i = 0; i < 4; ++i)
    for (int j = 0; j < 4; ++j) { a[i][j] = pm[i * 4 + j]; a[i][4 + j] = (i == j) ? 1.f : 0.f; }
  for (int c = 0; c < 4; ++c) {
    int piv = c; float best = fabsf(a[c][c]);
    for (int r = c + 1; r < 4; ++r) { float m = fabsf(a[r][c]); if (m > best) { best = m; piv = r; } }
    if (piv != c) for (int j = 0; j < 8; ++j) { float t = a[c][j]; a[c][j] = a[piv][j]; a[piv][j] = t; }
    float rd = 1.f / a[c][c];
    for (int j = 0; j < 8; ++j) a[c][j] *= rd;
    for (int r = 0; r < 4; ++r) if (r != c) {
      float f = a[r][c];
      for (int j = 0; j < 8; ++j) a[r][j] -= f * a[c][j];
    }
  }
  for (int v = 1; v < V; ++v) {
    const float* M = pm + v * 16;
    float* o = out + (v - 1) * 16;
    for (int i = 0; i < 4; ++i)
      for (int j = 0; j < 4; ++j) {
        float s = 0.f;
        for (int k = 0; k < 4; ++k) s += M[i * 4 + k] * a[k][4 + j];
        o[i * 4 + j] = s;
      }
  }
}

// ------------------ softmax over D, depth regression, conf -----------------
__global__ void umt_finalize_kernel(const float* __restrict__ costs, const float* __restrict__ depthv,
                                    float* __restrict__ out_depth, float* __restrict__ out_prob,
                                    float* __restrict__ out_conf, int D, int HW) {
  int p = blockIdx.x * blockDim.x + threadIdx.x;
  if (p >= HW) return;
  float pr[32];
  float m = -1e30f;
  for (int d = 0; d < D; ++d) m = fmaxf(m, costs[(size_t)d * HW + p]);
  float s = 0.f;
  for (int d = 0; d < D; ++d) { float e = expf(costs[(size_t)d * HW + p] - m); pr[d] = e; s += e; }
  float inv = 1.f / s;
  float dep = 0.f, ei = 0.f;
  for (int d = 0; d < D; ++d) {
    float q = pr[d] * inv;
    pr[d] = q;
    out_prob[(size_t)d * HW + p] = q;
    dep += q * depthv[d];
    ei  += q * (float)d;
  }
  out_depth[p] = dep;
  float eic = fminf(fmaxf(ei, 0.f), (float)(D - 1));
  int di = (int)eic;  // trunc toward zero, matches astype(int32) on nonneg
  float cf = 0.f;
  for (int t = di - 1; t <= di + 2; ++t)
    if (t >= 0 && t < D) cf += pr[t];
  out_conf[p] = cf;
}

// ===========================================================================
extern "C" void kernel_launch(void* const* d_in, const int* in_sizes, int n_in,
                              void* d_out, int out_size, void* d_ws, size_t ws_size,
                              hipStream_t stream) {
  (void)in_sizes; (void)n_in; (void)out_size; (void)ws_size;

  const int H = 512, W = 640, HF = 128, WF = 160, D = 32;
  const int HW0 = HF * WF;
  const int H1 = HF / 2, W1 = WF / 2, HW1 = H1 * W1;
  const int H2 = HF / 4, W2 = WF / 4, HW2 = H2 * W2;

  const float* imgs   = (const float*)d_in[0];
  const float* projm  = (const float*)d_in[1];
  const float* depthv = (const float*)d_in[2];
  // params: d_in[3 + 2*layer] = weight, d_in[4 + 2*layer] = bias, layers in
  // make_params order: f0..f5, g0..g2, c0..c4, u2, u3, o

  struct LDdef { int cin, cout, k; };
  static const LDdef L[17] = {
    {3, 16, 3}, {16, 16, 3}, {16, 32, 3}, {32, 32, 3}, {32, 32, 3}, {32, 32, 3},
    {32, 4, 1}, {4, 4, 1}, {4, 1, 1},
    {48, 64, 3}, {32, 64, 3}, {32, 64, 3}, {48, 64, 3}, {40, 32, 3},
    {16, 16, 3}, {16, 16, 3}, {8, 1, 3}
  };
  enum { Lf0, Lf1, Lf2, Lf3, Lf4, Lf5, Lg0, Lg1, Lg2,
         Lc0, Lc1, Lc2, Lc3, Lc4, Lu2, Lu3, Lo };

  // ----------------------------- workspace ---------------------------------
  char* base = (char*)d_ws;
  size_t off = 0;
  auto alloc = [&](size_t bytes) -> void* {
    void* p = base + off;
    off += (bytes + 255) & ~(size_t)255;
    return p;
  };

  half_t* pw[17];
  for (int l = 0; l < 17; ++l) {
    int K = L[l].cin * L[l].k * L[l].k;
    int KC = (K + 31) / 32, CT = (L[l].cout + 15) / 16;
    pw[l] = (half_t*)alloc((size_t)CT * KC * 512 * sizeof(half_t));
  }
  float* projbuf = (float*)alloc(32 * sizeof(float));
  float* feat[3];
  for (int v = 0; v < 3; ++v) feat[v] = (float*)alloc((size_t)32 * HW0 * sizeof(float));
  // hidden state (zeroed each call)
  float* h0h = (float*)alloc((size_t)16 * HW0 * 4);
  float* h0c = (float*)alloc((size_t)16 * HW0 * 4);
  float* h1h = (float*)alloc((size_t)16 * HW1 * 4);
  float* h1c = (float*)alloc((size_t)16 * HW1 * 4);
  float* h2h = (float*)alloc((size_t)16 * HW2 * 4);
  float* h2c = (float*)alloc((size_t)16 * HW2 * 4);
  float* h3h = (float*)alloc((size_t)16 * HW1 * 4);
  float* h3c = (float*)alloc((size_t)16 * HW1 * 4);
  float* h4h = (float*)alloc((size_t)8 * HW0 * 4);
  float* h4c = (float*)alloc((size_t)8 * HW0 * 4);
  float* costs = (float*)alloc((size_t)D * HW0 * 4);
  // scratch arena: featnet ping-pong, later reused for step temporaries
  float* arena = (float*)alloc((size_t)2 * 16 * H * W * sizeof(float));
  float* tmpA = arena;
  float* tmpB = arena + (size_t)16 * H * W;
  // per-step temps carved from the arena (featnet is finished by then)
  float* wvb = arena;
  float* acc = wvb + 32 * HW0;
  float* xin = acc + 32 * HW0;
  float* y0  = xin + 32 * HW0;
  float* y1t = y0 + 4 * HW0;
  float* y1  = y1t + 4 * HW0;
  float* rw  = y1 + 4 * HW0;
  float* cat = rw + HW0;
  float* zb  = cat + 48 * HW0;
  float* p1  = zb + 64 * HW0;
  float* p2  = p1 + 16 * HW1;
  float* ub  = p2 + 16 * HW2;
  float* u2b = ub + 16 * HW0;
  float* u3b = u2b + 16 * HW1;

  dim3 eb(EW_BLK);
  auto ew = [&](int n) { return dim3((unsigned)((n + EW_BLK - 1) / EW_BLK)); };

  // --------------------------- pack weights --------------------------------
  for (int l = 0; l < 17; ++l) {
    int K = L[l].cin * L[l].k * L[l].k;
    int KC = (K + 31) / 32, CT = (L[l].cout + 15) / 16;
    int tot = CT * KC * 256;
    umt_pack_w_kernel<<<ew(tot), eb, 0, stream>>>(
        (const float*)d_in[3 + 2 * l], pw[l], L[l].cin, L[l].cout, L[l].k);
  }
  umt_proj_prep_kernel<<<1, 1, 0, stream>>>(projm, projbuf, 3);

  // zero hidden-state span (h0h .. h4c, contiguous with padding)
  {
    size_t hidFloats = (size_t)((char*)costs - (char*)h0h) / 4;
    umt_zero_kernel<<<ew((int)hidFloats), eb, 0, stream>>>(h0h, (int)hidFloats);
  }

  // ------------------------------ helpers ----------------------------------
  auto conv = [&](const float* x, int li, float* y, int Hin, int Win, int stride) {
    int k = L[li].k, pad = k / 2;
    int Hout = (Hin + 2 * pad - k) / stride + 1;
    int Wout = (Win + 2 * pad - k) / stride + 1;
    int npix = Hout * Wout;
    int pt = (npix + 31) / 32;
    dim3 g((unsigned)((pt + 3) / 4), (unsigned)((L[li].cout + 15) / 16));
    const float* bias = (const float*)d_in[3 + 2 * li + 1];
    if (k == 3)
      umt_conv_wmma_kernel<3><<<g, 128, 0, stream>>>(
          x, pw[li], bias, y, L[li].cin, L[li].cout, Hin, Win, Hout, Wout, stride);
    else
      umt_conv_wmma_kernel<1><<<g, 128, 0, stream>>>(
          x, pw[li], bias, y, L[li].cin, L[li].cout, Hin, Win, Hout, Wout, stride);
  };
  auto gn = [&](float* x, int C, int HW) {
    umt_gn_relu_kernel<<<dim3(4), dim3(256), 0, stream>>>(x, x, C, HW, 4, 1e-5f);
  };
  auto cpy = [&](float* d, const float* s, int n) {
    umt_copy_kernel<<<ew(n), eb, 0, stream>>>(d, s, n);
  };

  // ------------------------------ featnet ----------------------------------
  for (int v = 0; v < 3; ++v) {
    const float* img = imgs + (size_t)v * 3 * H * W;
    conv(img,  Lf0, tmpA, H, W, 1);            gn(tmpA, 16, H * W);
    conv(tmpA, Lf1, tmpB, H, W, 1);            gn(tmpB, 16, H * W);
    conv(tmpB, Lf2, tmpA, H, W, 2);            gn(tmpA, 32, (H / 2) * (W / 2));
    conv(tmpA, Lf3, tmpB, H / 2, W / 2, 1);    gn(tmpB, 32, (H / 2) * (W / 2));
    conv(tmpB, Lf4, tmpA, H / 2, W / 2, 2);    gn(tmpA, 32, HW0);
    conv(tmpA, Lf5, feat[v], HF, WF, 1);
  }

  // --------------------------- recurrent loop ------------------------------
  for (int d = 0; d < D; ++d) {
    for (int s = 0; s < 2; ++s) {
      umt_warp_sq_kernel<<<ew(HW0), eb, 0, stream>>>(
          feat[1 + s], feat[0], projbuf + s * 16, depthv, d, wvb, 32, HF, WF);
      conv(wvb, Lg0, y0, HF, WF, 1);  gn(y0, 4, HW0);
      conv(y0, Lg1, y1t, HF, WF, 1);
      umt_add_relu_kernel<<<ew(4 * HW0), eb, 0, stream>>>(y1, y0, y1t, 4 * HW0);
      conv(y1, Lg2, rw, HF, WF, 1);
      umt_sigmoid_kernel<<<ew(HW0), eb, 0, stream>>>(rw, HW0);
      umt_acc_kernel<<<ew(32 * HW0), eb, 0, stream>>>(acc, wvb, rw, 32 * HW0, HW0, s == 0 ? 1 : 0);
    }
    umt_scale_kernel<<<ew(32 * HW0), eb, 0, stream>>>(xin, acc, -0.5f, 32 * HW0);

    // ---- UNet-ConvLSTM step ----
    cpy(cat, xin, 32 * HW0); cpy(cat + 32 * HW0, h0h, 16 * HW0);
    conv(cat, Lc0, zb, HF, WF, 1);
    umt_lstm_kernel<<<ew(16 * HW0), eb, 0, stream>>>(zb, h0h, h0c, 16, HW0);

    umt_maxpool2_kernel<<<ew(16 * HW1), eb, 0, stream>>>(h0h, p1, 16, HF, WF);
    cpy(cat, p1, 16 * HW1); cpy(cat + 16 * HW1, h1h, 16 * HW1);
    conv(cat, Lc1, zb, H1, W1, 1);
    umt_lstm_kernel<<<ew(16 * HW1), eb, 0, stream>>>(zb, h1h, h1c, 16, HW1);

    umt_maxpool2_kernel<<<ew(16 * HW2), eb, 0, stream>>>(h1h, p2, 16, H1, W1);
    cpy(cat, p2, 16 * HW2); cpy(cat + 16 * HW2, h2h, 16 * HW2);
    conv(cat, Lc2, zb, H2, W2, 1);
    umt_lstm_kernel<<<ew(16 * HW2), eb, 0, stream>>>(zb, h2h, h2c, 16, HW2);

    umt_up2_kernel<<<ew(16 * HW1), eb, 0, stream>>>(h2h, ub, 16, H2, W2);
    conv(ub, Lu2, u2b, H1, W1, 1);  gn(u2b, 16, HW1);
    cpy(cat, u2b, 16 * HW1); cpy(cat + 16 * HW1, h1h, 16 * HW1); cpy(cat + 32 * HW1, h3h, 16 * HW1);
    conv(cat, Lc3, zb, H1, W1, 1);
    umt_lstm_kernel<<<ew(16 * HW1), eb, 0, stream>>>(zb, h3h, h3c, 16, HW1);

    umt_up2_kernel<<<ew(16 * HW0), eb, 0, stream>>>(h3h, ub, 16, H1, W1);
    conv(ub, Lu3, u3b, HF, WF, 1);  gn(u3b, 16, HW0);
    cpy(cat, u3b, 16 * HW0); cpy(cat + 16 * HW0, h0h, 16 * HW0); cpy(cat + 32 * HW0, h4h, 8 * HW0);
    conv(cat, Lc4, zb, HF, WF, 1);
    umt_lstm_kernel<<<ew(8 * HW0), eb, 0, stream>>>(zb, h4h, h4c, 8, HW0);

    conv(h4h, Lo, costs + (size_t)d * HW0, HF, WF, 1);
  }

  // ------------------------------ finalize ---------------------------------
  float* out = (float*)d_out;
  umt_finalize_kernel<<<ew(HW0), eb, 0, stream>>>(
      costs, depthv, out, out + HW0, out + HW0 + (size_t)D * HW0, D, HW0);
}
